// WavCapsNet_44899588113034
// MI455X (gfx1250) — matrix operand order repaired and checked by reading, not compiled, not run on hardware
//
#include <hip/hip_runtime.h>
#include <hip/hip_fp16.h>

typedef __attribute__((ext_vector_type(16))) _Float16 v16h;
typedef __attribute__((ext_vector_type(8)))  float    v8f;

union Frag16 { v16h h; unsigned int u[8]; };

// CDNA5 16-bit A-matrix 16x32 layout: VGPR j (j<4) holds K=2j,2j+1 (+8 for lanes 16-31),
// VGPR j (j>=4) holds K=2j+8,2j+9 (+8 for lanes 16-31).
__device__ __forceinline__ int a_kbase(int j, int half) {
    return ((j < 4) ? (2 * j) : (2 * j + 8)) + (half ? 8 : 0);
}

// ---------------------------------------------------------------------------
// Kernel 1: generate wavelet weights (f16) + cast conv_w to f16
// ---------------------------------------------------------------------------
__global__ void __launch_bounds__(256) prep_weights(
    const float* __restrict__ a, const float* __restrict__ w,
    const float* __restrict__ conv_w,
    _Float16* __restrict__ wavw, _Float16* __restrict__ convw)
{
    int idx = blockIdx.x * 256 + threadIdx.x;
    if (idx < 128 * 512) {
        int f = idx >> 9, k = idx & 511;
        float t  = -1.0f + 2.0f * (float)k / 511.0f;
        float as = fmaxf(a[f], 1e-5f);
        float ts = t / as;
        float val = __cosf(w[f] * t) * __expf(-0.5f * ts * ts);
        wavw[idx] = (_Float16)val;
    }
    int idx2 = idx - 128 * 512;
    if (idx2 >= 0 && idx2 < 256 * 2048) {
        convw[idx2] = (_Float16)conv_w[idx2];
    }
}

// ---------------------------------------------------------------------------
// Kernel 2: WavConv1d as WMMA GEMM.
// Per block: 4 batches -> M=128 rows (output positions), N=128 filters, K=512.
// h[b][f][p] stored f16 (consumed as next GEMM's A operand).
// ---------------------------------------------------------------------------
#define XS 1016  // padded f16 stride for one x row (1008 valid)

__global__ void __launch_bounds__(256) wavconv_wmma(
    const float* __restrict__ x, const _Float16* __restrict__ wavw,
    _Float16* __restrict__ h)
{
    __shared__ _Float16 xs[4 * XS];      // 4 x-rows, f16
    __shared__ _Float16 wc[128 * 32];    // weight K-chunk: [filter][32]

    const int tid  = threadIdx.x;
    const int lane = tid & 31, wave = tid >> 5;
    const int waveM = wave & 3;          // 4 row-waves * 32 rows
    const int waveN = wave >> 2;         // 2 col-waves * 64 cols
    const int b0 = blockIdx.x * 4;
    const int m = lane & 15, half = lane >> 4;

    for (int i = tid; i < 4 * 1008; i += 256) {
        int bb = i / 1008, k = i % 1008;
        xs[bb * XS + k] = (_Float16)x[(size_t)(b0 + bb) * 1008 + k];
    }

    v8f acc[2][4];
    #pragma unroll
    for (int mt = 0; mt < 2; ++mt)
        #pragma unroll
        for (int nt = 0; nt < 4; ++nt)
            #pragma unroll
            for (int j = 0; j < 8; ++j) acc[mt][nt][j] = 0.0f;

    for (int kc = 0; kc < 16; ++kc) {
        __syncthreads();
        for (int i = tid; i < 4096; i += 256) {
            int f = i >> 5, kk = i & 31;
            wc[i] = wavw[f * 512 + kc * 32 + kk];
        }
        __syncthreads();

        Frag16 af[2];
        #pragma unroll
        for (int mt = 0; mt < 2; ++mt) {
            int row = waveM * 32 + mt * 16 + m;
            int bb = row >> 5, p = row & 31;
            const _Float16* xr = &xs[bb * XS + p * 16 + kc * 32];
            #pragma unroll
            for (int j = 0; j < 8; ++j)
                af[mt].u[j] = *(const unsigned int*)(xr + a_kbase(j, half));
        }
        #pragma unroll
        for (int nt = 0; nt < 4; ++nt) {
            Frag16 bf;
            const _Float16* wr = &wc[(waveN * 64 + nt * 16 + m) * 32];
            #pragma unroll
            for (int j = 0; j < 8; ++j)
                bf.u[j] = *(const unsigned int*)(wr + a_kbase(j, half));
            #pragma unroll
            for (int mt = 0; mt < 2; ++mt)
                acc[mt][nt] = __builtin_amdgcn_wmma_f32_16x16x32_f16(
                    false, af[mt].h, false, bf.h, (short)0, acc[mt][nt], false, false);
        }
    }

    #pragma unroll
    for (int mt = 0; mt < 2; ++mt) {
        int rowbase = waveM * 32 + mt * 16 + half * 8;
        #pragma unroll
        for (int nt = 0; nt < 4; ++nt) {
            int f = waveN * 64 + nt * 16 + m;
            #pragma unroll
            for (int j = 0; j < 8; ++j) {
                int row = rowbase + j;
                int bb = row >> 5, p = row & 31;
                h[(size_t)(b0 + bb) * 4096 + f * 32 + p] = (_Float16)acc[mt][nt][j];
            }
        }
    }
}

// ---------------------------------------------------------------------------
// Kernel 3: PrimaryCaps conv as implicit-im2col WMMA GEMM.
// M = 4096*9 rows (b,q), N = 256 out-channels (2 col-blocks of 128), K = 2048.
// A[row][K] = h[b][K/16][2q + K%16]; pairs (K,K+1) are contiguous in h.
// ---------------------------------------------------------------------------
__global__ void __launch_bounds__(256) pcaps_wmma(
    const _Float16* __restrict__ h, const _Float16* __restrict__ convw,
    const float* __restrict__ conv_b, float* __restrict__ p)
{
    __shared__ _Float16 wc[128 * 32];

    const int tid  = threadIdx.x;
    const int lane = tid & 31, wave = tid >> 5;
    const int waveM = wave & 3;
    const int waveN = wave >> 2;
    const int r0 = blockIdx.x * 128;     // 288 row-blocks
    const int n0 = blockIdx.y * 128;     // 2 col-blocks
    const int m = lane & 15, half = lane >> 4;

    v8f acc[2][4];
    #pragma unroll
    for (int mt = 0; mt < 2; ++mt)
        #pragma unroll
        for (int nt = 0; nt < 4; ++nt)
            #pragma unroll
            for (int j = 0; j < 8; ++j) acc[mt][nt][j] = 0.0f;

    int bb[2], q2[2];
    #pragma unroll
    for (int mt = 0; mt < 2; ++mt) {
        int row = r0 + waveM * 32 + mt * 16 + m;
        bb[mt] = row / 9;
        q2[mt] = 2 * (row % 9);
    }

    for (int kc = 0; kc < 64; ++kc) {
        __syncthreads();
        for (int i = tid; i < 4096; i += 256) {
            int n = i >> 5, kk = i & 31;
            wc[i] = convw[(size_t)(n0 + n) * 2048 + kc * 32 + kk];
        }
        if (kc + 1 < 64)  // software prefetch of next weight chunk (global_prefetch_b8)
            __builtin_prefetch(convw + (size_t)(n0 + (tid & 127)) * 2048 + (kc + 1) * 32, 0, 1);
        __syncthreads();

        Frag16 af[2];
        #pragma unroll
        for (int mt = 0; mt < 2; ++mt) {
            const _Float16* hb = h + (size_t)bb[mt] * 4096;
            #pragma unroll
            for (int j = 0; j < 8; ++j) {
                int K = kc * 32 + a_kbase(j, half);
                int c = K >> 4, k2 = K & 15;
                af[mt].u[j] = *(const unsigned int*)(hb + c * 32 + q2[mt] + k2);
            }
        }
        #pragma unroll
        for (int nt = 0; nt < 4; ++nt) {
            Frag16 bf;
            const _Float16* wr = &wc[(waveN * 64 + nt * 16 + m) * 32];
            #pragma unroll
            for (int j = 0; j < 8; ++j)
                bf.u[j] = *(const unsigned int*)(wr + a_kbase(j, half));
            #pragma unroll
            for (int mt = 0; mt < 2; ++mt)
                acc[mt][nt] = __builtin_amdgcn_wmma_f32_16x16x32_f16(
                    false, af[mt].h, false, bf.h, (short)0, acc[mt][nt], false, false);
        }
    }

    #pragma unroll
    for (int mt = 0; mt < 2; ++mt) {
        int rowbase = r0 + waveM * 32 + mt * 16 + half * 8;
        #pragma unroll
        for (int nt = 0; nt < 4; ++nt) {
            int o = n0 + waveN * 64 + nt * 16 + m;
            float bias = conv_b[o];
            #pragma unroll
            for (int j = 0; j < 8; ++j) {
                int row = rowbase + j;
                int b = row / 9, q = row % 9;
                p[(size_t)b * 2304 + o * 9 + q] = acc[mt][nt][j] + bias;
            }
        }
    }
}

// ---------------------------------------------------------------------------
// Kernel 4: squash + u_hat + 3-iteration dynamic routing. One block per batch,
// 288 threads (thread c owns capsule c). s_j reduced with LDS float atomics.
// ---------------------------------------------------------------------------
__global__ void __launch_bounds__(288) routing_kernel(
    const float* __restrict__ p, const float* __restrict__ Wc,
    float* __restrict__ out_probs, float* __restrict__ out_c)
{
    __shared__ float sj[64];
    __shared__ float vj[64];

    const int b = blockIdx.x;
    const int c = threadIdx.x;  // 0..287

    // squash(u) over the 8-vector of capsule c
    float u[8];
    float sn = 0.0f;
    const float* pb = p + (size_t)b * 2304 + c * 8;
    #pragma unroll
    for (int i = 0; i < 8; ++i) { u[i] = pb[i]; sn += u[i] * u[i]; }
    float scale = (sn / (1.0f + sn)) * rsqrtf(sn + 1e-8f);
    #pragma unroll
    for (int i = 0; i < 8; ++i) u[i] *= scale;

    // u_hat[d][o] = sum_i W_caps[c,d,o,i] * u[i]   (64 values in registers)
    float uh[64];
    const float* Wcc = Wc + (size_t)c * 512;
    #pragma unroll
    for (int t = 0; t < 64; ++t) {
        const float* wr = Wcc + t * 8;
        float s = 0.0f;
        #pragma unroll
        for (int i = 0; i < 8; ++i) s += wr[i] * u[i];
        uh[t] = s;
    }

    float bij[4] = {0.0f, 0.0f, 0.0f, 0.0f};
    float cij[4];

    for (int it = 0; it < 3; ++it) {
        // softmax over the 4 classes
        float mx = fmaxf(fmaxf(bij[0], bij[1]), fmaxf(bij[2], bij[3]));
        float es = 0.0f;
        #pragma unroll
        for (int d = 0; d < 4; ++d) { cij[d] = __expf(bij[d] - mx); es += cij[d]; }
        float inv = 1.0f / es;
        #pragma unroll
        for (int d = 0; d < 4; ++d) cij[d] *= inv;

        // s_j[d][o] = sum_c cij[d]*uh[d][o]  via LDS atomics (ds_add_f32)
        __syncthreads();
        if (c < 64) sj[c] = 0.0f;
        __syncthreads();
        #pragma unroll
        for (int t = 0; t < 64; ++t)
            atomicAdd(&sj[t], cij[t >> 4] * uh[t]);
        __syncthreads();

        // v = squash(s) per class
        if (c < 4) {
            float s2 = 0.0f;
            #pragma unroll
            for (int o = 0; o < 16; ++o) { float s = sj[c * 16 + o]; s2 += s * s; }
            float sc = (s2 / (1.0f + s2)) * rsqrtf(s2 + 1e-8f);
            #pragma unroll
            for (int o = 0; o < 16; ++o) vj[c * 16 + o] = sj[c * 16 + o] * sc;
        }
        __syncthreads();

        if (it < 2) {
            #pragma unroll
            for (int d = 0; d < 4; ++d) {
                float ag = 0.0f;
                #pragma unroll
                for (int o = 0; o < 16; ++o) ag += uh[d * 16 + o] * vj[d * 16 + o];
                bij[d] += ag;
            }
        }
    }

    if (c < 4) {
        float s2 = 0.0f;
        #pragma unroll
        for (int o = 0; o < 16; ++o) { float v = vj[c * 16 + o]; s2 += v * v; }
        out_probs[(size_t)b * 4 + c] = sqrtf(s2);
    }
    #pragma unroll
    for (int d = 0; d < 4; ++d)
        out_c[((size_t)b * 288 + c) * 4 + d] = cij[d];
}

// ---------------------------------------------------------------------------
extern "C" void kernel_launch(void* const* d_in, const int* in_sizes, int n_in,
                              void* d_out, int out_size, void* d_ws, size_t ws_size,
                              hipStream_t stream)
{
    const float* x      = (const float*)d_in[0];  // [4096,1,1008]
    const float* a      = (const float*)d_in[1];  // [128]
    const float* w      = (const float*)d_in[2];  // [128]
    const float* conv_w = (const float*)d_in[3];  // [256,128,16]
    const float* conv_b = (const float*)d_in[4];  // [256]
    const float* Wcaps  = (const float*)d_in[5];  // [288,4,16,8]
    float* out = (float*)d_out;                   // [4096*4] probs ++ [4096*288*4] c_ij

    char* ws = (char*)d_ws;
    _Float16* wavw  = (_Float16*)(ws + 0);         //  128*512  f16 = 128 KB
    _Float16* convw = (_Float16*)(ws + 131072);    //  256*2048 f16 =   1 MB
    _Float16* hbuf  = (_Float16*)(ws + 1179648);   // 4096*128*32 f16 = 32 MB
    float*    pbuf  = (float*)   (ws + 34734080);  // 4096*2304  f32 = 36 MB

    prep_weights <<<2304, 256, 0, stream>>>(a, w, conv_w, wavw, convw);
    wavconv_wmma <<<1024, 256, 0, stream>>>(x, wavw, hbuf);
    pcaps_wmma   <<<dim3(288, 2), 256, 0, stream>>>(hbuf, convw, conv_b, pbuf);
    routing_kernel<<<4096, 288, 0, stream>>>(pbuf, Wcaps, out, out + 16384);
}
